// Qwen3MoE_34806414967305
// MI455X (gfx1250) — compile-verified, matrix operand order checked by reading, not run on hardware
//
#include <hip/hip_runtime.h>
#include <hip/hip_bf16.h>
#include <cstddef>
#include <cstdint>

// ---------------- problem constants ----------------
constexpr int Tt   = 2048;   // B*S tokens
constexpr int Hd   = 1024;   // hidden
constexpr int Id   = 1536;   // intermediate
constexpr int Ee   = 32;     // experts
constexpr int TOPK = 8;
constexpr int IC   = 128;            // I-chunk width
constexpr int NCH  = Id / IC;        // 12 chunks
constexpr int TILE_M = 16;           // tokens per block tile
constexpr int BLOCK  = 256;          // 8 waves of 32

typedef __attribute__((ext_vector_type(16))) __bf16        v16bf;
typedef __attribute__((ext_vector_type(8)))  float         v8f;
typedef __attribute__((ext_vector_type(4)))  unsigned int  v4u;
typedef __attribute__((ext_vector_type(8)))  int           v8i;
typedef __attribute__((ext_vector_type(4)))  int           v4i;

#if defined(__HIP_DEVICE_COMPILE__) && __has_builtin(__builtin_amdgcn_tensor_load_to_lds) && \
    __has_builtin(__builtin_amdgcn_s_wait_tensorcnt)
#define USE_TDM 1
#else
#define USE_TDM 0
#endif

// ---------------- fast math ----------------
__device__ __forceinline__ float fast_rcp(float x) { return __builtin_amdgcn_rcpf(x); }
__device__ __forceinline__ float silu_f(float g) { return g * fast_rcp(1.0f + __expf(-g)); }
__device__ __forceinline__ float sigmoid_f(float g) { return fast_rcp(1.0f + __expf(-g)); }

// ---------------- WMMA fragment helpers ----------------
// A (16x32 bf16, MxK): lane<16 -> row M=lane, elems 0..7 = K(k0+0..7), 8..15 = K(k0+16..23)
//                      lane>=16 -> row M=lane-16, elems 0..7 = K(k0+8..15), 8..15 = K(k0+24..31)
__device__ __forceinline__ v16bf load_a_frag(const __bf16* row, int k0, int half) {
    v16bf a;
    const __bf16* p = row + k0 + half * 8;
#pragma unroll
    for (int j = 0; j < 8; ++j) { a[j] = p[j]; a[8 + j] = p[16 + j]; }
    return a;
}

// B (32x16 bf16, KxN) mirrored: lane -> column N, same K-half split.
// Source is a contiguous fp32 row of the weight matrix (W[n*ld + k]), converted to bf16.
__device__ __forceinline__ v16bf load_b_frag_f32(const float* col, int k0, int half) {
    v16bf b;
    const float* p = col + k0 + half * 8;
#pragma unroll
    for (int j = 0; j < 8; ++j) { b[j] = (__bf16)p[j]; b[8 + j] = (__bf16)p[16 + j]; }
    return b;
}

#if USE_TDM
// ---------------- Tensor Data Mover: gather 16 token rows of xb into LDS ----------------
// D# per CDNA5 ISA, gather mode, 16-bit indices.
// 6-arg builtin form: (v4u g0, v8i g1, v4i g2, v4i g3, v8i g4, i32 cpol)
__device__ __forceinline__ void tdm_gather_x(const __bf16* xb, unsigned lds_addr,
                                             const int* toks /* 16 row indices */)
{
    unsigned long long ga = (unsigned long long)(uintptr_t)xb;
    v4u g0;
    g0[0] = 0x80000001u;                                  // count=1, gather_mode=1, idx16
    g0[1] = lds_addr;                                     // LDS byte address
    g0[2] = (unsigned)(ga & 0xFFFFFFFFu);                 // global_addr[31:0]
    g0[3] = (unsigned)((ga >> 32) & 0x01FFFFFFu) | (2u << 30); // addr[56:32], type=2
    v8i g1;
    g1[0] = 0x00010000;                                   // data_size = 1 (2 bytes)
    g1[1] = (int)((Hd & 0xFFFF) << 16);                   // tensor_dim0[15:0]
    g1[2] = (int)((Hd >> 16) | ((Tt & 0xFFFF) << 16));    // tensor_dim0 hi | tensor_dim1 lo
    g1[3] = (int)((Tt >> 16) | ((Hd & 0xFFFF) << 16));    // tensor_dim1 hi | tile_dim0 = Hd
    g1[4] = TILE_M;                                       // tile_dim1 = #valid indices (16)
    g1[5] = Hd;                                           // tensor_dim0_stride[31:0]
    g1[6] = 0;
    g1[7] = 0;
    v4i g2, g3;
#pragma unroll
    for (int j = 0; j < 4; ++j) {
        g2[j] = (toks[2 * j]     & 0xFFFF) | (toks[2 * j + 1]     << 16);
        g3[j] = (toks[8 + 2 * j] & 0xFFFF) | (toks[8 + 2 * j + 1] << 16);
    }
    v8i g4 = {};                                          // unused trailing group
    __builtin_amdgcn_tensor_load_to_lds(g0, g1, g2, g3, g4, 0);
}
#endif

// ---------------- shared tile-MLP body ----------------
// Computes, for 16 tokens (toks/wts in LDS):
//   y = (silu(x@Gw^T) * (x@Uw^T)) @ Dw^T ;  out[tok] += wts * y   (atomic)
// Gw,Uw: [I,H] fp32 row-major.  Dw: [H,I] fp32 row-major.  xb: [T,H] bf16.
__device__ __forceinline__ void tile_mlp_body(
    const __bf16* __restrict__ xb,
    const float* __restrict__ gw, const float* __restrict__ uw,
    const float* __restrict__ dw,
    __bf16* xs,           // LDS [TILE_M * Hd]
    __bf16* act_s,        // LDS [TILE_M * IC]
    const int* toks, const float* wts,   // LDS [TILE_M]
    float* __restrict__ out)
{
    const int tid  = threadIdx.x;
    const int wave = tid >> 5;
    const int lane = tid & 31;
    const int half = lane >> 4;
    const int lm   = lane & 15;

#if USE_TDM
    // single TDM gather op stages the whole 16x1024 bf16 tile into LDS
    if (wave == 0) {
        tdm_gather_x(xb, (unsigned)(size_t)xs, toks);
        __builtin_amdgcn_s_wait_tensorcnt(0);
    }
    __syncthreads();
#else
    {
        uint4* dst = (uint4*)xs;
        for (int i = tid; i < TILE_M * (Hd / 8); i += BLOCK) {
            int r = i >> 7, c = i & 127;
            const uint4* srow = (const uint4*)(xb + (size_t)toks[r] * Hd);
            dst[r * (Hd / 8) + c] = srow[c];
        }
    }
    __syncthreads();
#endif

    // register-resident down-proj accumulators: this wave owns y[16, wave*128 .. +127]
    v8f yacc[8];
#pragma unroll
    for (int s = 0; s < 8; ++s) yacc[s] = {};

    const __bf16* xrow = xs + lm * Hd;     // A row for this lane (token lm)

    for (int ch = 0; ch < NCH; ++ch) {
        const int i0 = ch * IC;

        // ---- gate & up: this wave's 16 I-columns, K = Hd ----
        const int    ni   = i0 + wave * 16 + lm;         // absolute I column for this lane
        const float* gcol = gw + (size_t)ni * Hd;
        const float* ucol = uw + (size_t)ni * Hd;
        if (ch + 1 < NCH) {                              // keep HBM weight stream ahead
            __builtin_prefetch(gcol + (size_t)IC * Hd, 0, 0);
            __builtin_prefetch(ucol + (size_t)IC * Hd, 0, 0);
        }
        v8f gacc = {}, uacc = {};
        for (int k0 = 0; k0 < Hd; k0 += 32) {
            v16bf a  = load_a_frag(xrow, k0, half);
            v16bf bg = load_b_frag_f32(gcol, k0, half);
            v16bf bu = load_b_frag_f32(ucol, k0, half);
            gacc = __builtin_amdgcn_wmma_f32_16x16x32_bf16(false, a, false, bg,
                                                           (short)0, gacc, false, false);
            uacc = __builtin_amdgcn_wmma_f32_16x16x32_bf16(false, a, false, bu,
                                                           (short)0, uacc, false, false);
        }
        // act = silu(g)*u -> bf16 LDS  (C/D layout: VGPR r holds row r + 8*half, col lm)
#pragma unroll
        for (int r = 0; r < 8; ++r) {
            float av = silu_f(gacc[r]) * uacc[r];
            int   m  = r + half * 8;
            act_s[m * IC + wave * 16 + lm] = (__bf16)av;
        }
        __syncthreads();

        // ---- down-proj partial: yacc += act[16,128] @ Dw[:, i0:i0+128]^T ----
        const __bf16* arow = act_s + lm * IC;
        v16bf afr[4];
#pragma unroll
        for (int ks = 0; ks < 4; ++ks) afr[ks] = load_a_frag(arow, ks * 32, half);

#pragma unroll
        for (int s2 = 0; s2 < 8; ++s2) {
            const int    nh   = wave * 128 + s2 * 16 + lm;   // output H column
            const float* dcol = dw + (size_t)nh * Id + i0;
#pragma unroll
            for (int ks = 0; ks < 4; ++ks) {
                v16bf b = load_b_frag_f32(dcol, ks * 32, half);
                yacc[s2] = __builtin_amdgcn_wmma_f32_16x16x32_bf16(false, afr[ks], false, b,
                                                                   (short)0, yacc[s2], false, false);
            }
        }
        __syncthreads();
    }

    // scaled atomic accumulate into output
#pragma unroll
    for (int s2 = 0; s2 < 8; ++s2) {
#pragma unroll
        for (int r = 0; r < 8; ++r) {
            int   m = r + half * 8;
            int   n = wave * 128 + s2 * 16 + lm;
            float v = wts[m] * yacc[s2][r];
            atomicAdd(&out[(size_t)toks[m] * Hd + n], v);
        }
    }
}

// ---------------- kernels ----------------

// one wave per token; lane = expert. softmax + top-8 + gather lists + usage.
__global__ __launch_bounds__(BLOCK)
void router_kernel(const float* __restrict__ x, const float* __restrict__ gate_w,
                   int* __restrict__ counts, float* __restrict__ usage,
                   int* __restrict__ lists, float* __restrict__ listw)
{
    const int wave = threadIdx.x >> 5;
    const int lane = threadIdx.x & 31;
    const int t = blockIdx.x * 8 + wave;
    if (t >= Tt) return;

    const float* xr = x + (size_t)t * Hd;
    const float* wr = gate_w + (size_t)lane * Hd;
    float acc = 0.f;
    for (int h = 0; h < Hd; h += 4) {
        float4 xv = *(const float4*)(xr + h);
        float4 wv = *(const float4*)(wr + h);
        acc += xv.x * wv.x + xv.y * wv.y + xv.z * wv.z + xv.w * wv.w;
    }
    // wave-wide softmax over 32 experts
    float m = acc;
    for (int off = 16; off; off >>= 1) m = fmaxf(m, __shfl_xor(m, off, 32));
    float ex = __expf(acc - m);
    float s = ex;
    for (int off = 16; off; off >>= 1) s += __shfl_xor(s, off, 32);
    float prob = ex * fast_rcp(s);

    atomicAdd(&usage[lane], prob);

    // iterative top-8 (argmax with lane tie-break)
    float v = prob;
    for (int k = 0; k < TOPK; ++k) {
        float mv = v; int ml = lane;
        for (int off = 16; off; off >>= 1) {
            float ov = __shfl_xor(mv, off, 32);
            int   ol = __shfl_xor(ml, off, 32);
            if (ov > mv || (ov == mv && ol < ml)) { mv = ov; ml = ol; }
        }
        if (lane == ml) {
            int pos = atomicAdd(&counts[lane], 1);
            lists[lane * Tt + pos] = t;
            listw[lane * Tt + pos] = prob;
            v = -1.0f;
        }
    }
}

__global__ __launch_bounds__(BLOCK)
void cvt_kernel(const float* __restrict__ x, __bf16* __restrict__ xb)
{
    int i = blockIdx.x * blockDim.x + threadIdx.x;
    int base = i * 4;
    if (base >= Tt * Hd) return;
    float4 v = *(const float4*)(x + base);
    xb[base + 0] = (__bf16)v.x;
    xb[base + 1] = (__bf16)v.y;
    xb[base + 2] = (__bf16)v.z;
    xb[base + 3] = (__bf16)v.w;
}

__global__ __launch_bounds__(BLOCK)
void expert_kernel(const __bf16* __restrict__ xb,
                   const float* __restrict__ egw, const float* __restrict__ euw,
                   const float* __restrict__ edw,
                   const int* __restrict__ counts, const int* __restrict__ lists,
                   const float* __restrict__ listw,
                   float* __restrict__ out)
{
    __shared__ __bf16 xs[TILE_M * Hd];
    __shared__ __bf16 act_s[TILE_M * IC];
    __shared__ int    toks[TILE_M];
    __shared__ float  wts[TILE_M];

    const int e    = blockIdx.x;
    const int tile = blockIdx.y;
    const int cnt  = counts[e];
    if (tile * TILE_M >= cnt) return;   // uniform per block

    if (threadIdx.x < TILE_M) {
        int idx = tile * TILE_M + threadIdx.x;
        if (idx < cnt) {
            toks[threadIdx.x] = lists[e * Tt + idx];
            wts[threadIdx.x]  = listw[e * Tt + idx];
        } else {
            toks[threadIdx.x] = 0;
            wts[threadIdx.x]  = 0.f;   // padded rows contribute nothing
        }
    }
    __syncthreads();

    tile_mlp_body(xb,
                  egw + (size_t)e * Id * Hd,
                  euw + (size_t)e * Id * Hd,
                  edw + (size_t)e * Hd * Id,
                  xs, act_s, toks, wts, out);
}

__global__ __launch_bounds__(BLOCK)
void shared_kernel(const float* __restrict__ x, const __bf16* __restrict__ xb,
                   const float* __restrict__ sgw, const float* __restrict__ suw,
                   const float* __restrict__ sdw, const float* __restrict__ seg,
                   float* __restrict__ out)
{
    __shared__ __bf16 xs[TILE_M * Hd];
    __shared__ __bf16 act_s[TILE_M * IC];
    __shared__ int    toks[TILE_M];
    __shared__ float  wts[TILE_M];

    const int t0   = blockIdx.x * TILE_M;
    const int wave = threadIdx.x >> 5;
    const int lane = threadIdx.x & 31;

    // per-token sigmoid gate, two tokens per wave
    for (int j = wave; j < TILE_M; j += 8) {
        const int t = t0 + j;
        const float* xr = x + (size_t)t * Hd;
        float acc = 0.f;
        for (int h = lane; h < Hd; h += 32) acc += xr[h] * seg[h];
        for (int off = 16; off; off >>= 1) acc += __shfl_xor(acc, off, 32);
        if (lane == 0) {
            toks[j] = t;
            wts[j]  = sigmoid_f(acc);
        }
    }
    __syncthreads();

    tile_mlp_body(xb, sgw, suw, sdw, xs, act_s, toks, wts, out);
}

__global__ void aux_kernel(const float* __restrict__ usage, float* __restrict__ out_aux)
{
    const int lane = threadIdx.x;
    float u = usage[lane] * (1.0f / (float)Tt);
    float d = u - 1.0f / (float)Ee;
    float v = d * d;
    for (int off = 16; off; off >>= 1) v += __shfl_xor(v, off, 32);
    if (lane == 0) out_aux[0] = v * (1.0f / (float)Ee);
}

// ---------------- launch ----------------
extern "C" void kernel_launch(void* const* d_in, const int* in_sizes, int n_in,
                              void* d_out, int out_size, void* d_ws, size_t ws_size,
                              hipStream_t stream)
{
    const float* x    = (const float*)d_in[0];
    const float* gw   = (const float*)d_in[1];
    const float* egw  = (const float*)d_in[2];
    const float* euw  = (const float*)d_in[3];
    const float* edw  = (const float*)d_in[4];
    const float* sgw  = (const float*)d_in[5];
    const float* suw  = (const float*)d_in[6];
    const float* sdw  = (const float*)d_in[7];
    const float* seg  = (const float*)d_in[8];
    float* out = (float*)d_out;

    char* ws = (char*)d_ws;
    size_t off = 0;
    __bf16* xb     = (__bf16*)(ws + off); off += (size_t)Tt * Hd * 2;   // 4 MB
    int*    counts = (int*)  (ws + off); off += 128;
    float*  usage  = (float*)(ws + off); off += 128;
    int*    lists  = (int*)  (ws + off); off += (size_t)Ee * Tt * 4;    // 256 KB
    float*  listw  = (float*)(ws + off); off += (size_t)Ee * Tt * 4;    // 256 KB

    (void)hipMemsetAsync(out, 0, (size_t)out_size * sizeof(float), stream);
    (void)hipMemsetAsync(counts, 0, 256, stream);   // counts + usage

    router_kernel<<<Tt / 8, BLOCK, 0, stream>>>(x, gw, counts, usage, lists, listw);
    cvt_kernel<<<(Tt * Hd / 4 + BLOCK - 1) / BLOCK, BLOCK, 0, stream>>>(x, xb);

    dim3 ge(Ee, Tt / TILE_M);
    expert_kernel<<<ge, BLOCK, 0, stream>>>(xb, egw, euw, edw, counts, lists, listw, out);
    shared_kernel<<<Tt / TILE_M, BLOCK, 0, stream>>>(x, xb, sgw, suw, sdw, seg, out);
    aux_kernel<<<1, 32, 0, stream>>>(usage, out + (size_t)out_size - 1);
}
